// OneFormerPixelDecoder_83760452207142
// MI455X (gfx1250) — matrix operand order repaired and checked by reading, not compiled, not run on hardware
//
#include <hip/hip_runtime.h>

// ---------------- problem constants ----------------
#define EMBED   256
#define HEADS   8
#define CH      32          // EMBED / HEADS
#define LEVELS  3
#define POINTS  4
#define NQ      21504       // sum of level h*w
#define BATCH   4
#define MROWS   (BATCH * NQ)   // 86016 (divisible by 128)
#define NOA     288            // 192 offsets + 96 attn logits
#define NOAP    320            // padded to a multiple of 64 for the GEMM tiler

// ---------------- bf16 helpers (bit-level, RNE) ----------------
__device__ __forceinline__ unsigned short f2bf(float f) {
    union { float f; unsigned u; } v; v.f = f;
    unsigned r = v.u + 0x7FFFu + ((v.u >> 16) & 1u);
    return (unsigned short)(r >> 16);
}
__device__ __forceinline__ float bf2f(unsigned short u) {
    union { unsigned u; float f; } v; v.u = ((unsigned)u) << 16;
    return v.f;
}

// ---------------- WMMA types ----------------
typedef __attribute__((ext_vector_type(16))) __bf16 v16bf;
typedef __attribute__((ext_vector_type(8)))  float  v8f;

union FragAB { uint4 u[2]; v16bf bf; };

// ---------------- CDNA5 async global->LDS (ASYNCcnt path) ----------------
// the LDS destination address comes from the VDST operand; low 32 bits of a flat shared pointer are the
// LDS byte address (ISA 10.2 aperture mapping), so we pass that as VDST.
__device__ __forceinline__ void async_copy_b128(const void* gaddr, void* lds) {
    unsigned l32 = (unsigned)(size_t)lds;
    asm volatile("global_load_async_to_lds_b128 %0, %1, off"
                 :: "v"(l32), "v"(gaddr) : "memory");
}
__device__ __forceinline__ void async_wait0() {
    asm volatile("s_wait_asynccnt 0x0" ::: "memory");
}

// =====================================================================
// Prep kernels
// =====================================================================
__global__ __launch_bounds__(256) void prep_acts(
    const float* __restrict__ hs, const float* __restrict__ pos,
    unsigned short* __restrict__ hsb, unsigned short* __restrict__ qb, int n)
{
    int i = blockIdx.x * 256 + threadIdx.x;
    if (i < n) {
        float h = hs[i];
        hsb[i] = f2bf(h);
        qb[i]  = f2bf(h + pos[i]);
    }
}

// Weights converted to bf16 AND transposed to [N][K] so GEMM tiles are
// straight contiguous copies (async-to-LDS friendly, no in-LDS transpose).
__global__ __launch_bounds__(256) void prep_weights(
    const float* __restrict__ Wv,    const float* __restrict__ Woff,
    const float* __restrict__ Wattn, const float* __restrict__ Wout,
    const float* __restrict__ boff,  const float* __restrict__ battn,
    unsigned short* __restrict__ wvT,   // [256n][256k]
    unsigned short* __restrict__ woaT,  // [320n][256k] (rows 288..319 zero)
    unsigned short* __restrict__ woutT, // [256n][256k]
    float* __restrict__ boa)            // [320]
{
    int i = blockIdx.x * 256 + threadIdx.x;   // i = n*256 + k
    if (i < NOAP * EMBED) {
        int n = i / EMBED, k = i % EMBED;
        if (n < EMBED) {
            wvT[i]   = f2bf(Wv[k * EMBED + n]);
            woutT[i] = f2bf(Wout[k * EMBED + n]);
        }
        float v = (n < 192) ? Woff[k * 192 + n]
                : (n < NOA) ? Wattn[k * 96 + (n - 192)]
                : 0.f;
        woaT[i] = f2bf(v);
    }
    if (i < NOAP) {
        boa[i] = (i < 192) ? boff[i] : (i < NOA) ? battn[i - 192] : 0.f;
    }
}

// =====================================================================
// bf16 WMMA GEMM: C[M,N] = A[M,K] * Bt[N,K]^T + bias
// block = 256 threads = 8 waves; block tile 128(M) x 64(N); K staged by 32
// with double-buffered LDS filled by async global->LDS copies.
// Waves: 4(M) x 2(N); each wave owns a 2x2 grid of 16x16 WMMA tiles
// -> 4 v_wmma per K-step per wave, one barrier per K-step.
// Main loop issues the next tile unconditionally; last K-step is peeled.
// =====================================================================
#define TM 128
#define TN 64
#define TK 32

template <bool OUT_BF16>
__global__ __launch_bounds__(256) void gemm_bf16_wmma(
    const unsigned short* __restrict__ A,   // [M,K] bf16 row-major
    const unsigned short* __restrict__ Bt,  // [N,K] bf16 (transposed weights)
    const float* __restrict__ bias,         // [N]
    void* __restrict__ Cout,                // [M,N] f32 or bf16
    int M, int N, int K)
{
    __shared__ unsigned short lA[2][TM * TK];   // [m][k] row-major, 2x8KB
    __shared__ unsigned short lB[2][TN * TK];   // [n][k] row-major, 2x4KB

    const int tid  = threadIdx.x;
    const int lane = tid & 31;
    const int wave = tid >> 5;
    const int wm   = wave & 3;     // M sub-block (32 rows each)
    const int wn   = wave >> 2;    // N sub-block (32 cols each)
    const int rowBase = blockIdx.y * TM;
    const int colBase = blockIdx.x * TN;

    v8f acc[2][2];
#pragma unroll
    for (int i = 0; i < 2; ++i)
#pragma unroll
        for (int j = 0; j < 2; ++j)
            acc[i][j] = (v8f){0.f,0.f,0.f,0.f,0.f,0.f,0.f,0.f};

    // per-thread async-copy coordinates (straight 16B transfers)
    const int ar = tid >> 1;            // 0..127
    const int ak = (tid & 1) * 16;      // 0 or 16
    const int bn = tid >> 2;            // 0..63
    const int bk = (tid & 3) * 8;       // 0,8,16,24

    const int hf = lane >> 4;           // K-half select for fragments
    const int ml = lane & 15;

    const unsigned short* gA = A  + (size_t)(rowBase + ar) * K + ak;
    const unsigned short* gB = Bt + (size_t)(colBase + bn) * K + bk;

    auto issue_tile = [&](int k0, int buf) {
        async_copy_b128(gA + k0,     &lA[buf][ar * TK + ak]);
        async_copy_b128(gA + k0 + 8, &lA[buf][ar * TK + ak + 8]);
        async_copy_b128(gB + k0,     &lB[buf][bn * TK + bk]);
    };

    // ---- fragment gather + 4 WMMAs for one staged K-tile ----
    // A 16x32: elems 0..7 -> K = 8*hf + e; elems 8..15 -> K = 16+8*hf+(e-8)
    // B 32x16: elems 0..7 -> K = 8*hf + e; elems 8..15 -> K = 16+8*hf+(e-8)
    auto compute_tile = [&](int buf) {
        FragAB af[2], bf[2];
#pragma unroll
        for (int i = 0; i < 2; ++i) {
            const int mr = wm * 32 + i * 16 + ml;
            af[i].u[0] = *(const uint4*)&lA[buf][mr * TK + 8 * hf];
            af[i].u[1] = *(const uint4*)&lA[buf][mr * TK + 16 + 8 * hf];
        }
#pragma unroll
        for (int j = 0; j < 2; ++j) {
            const int nc = wn * 32 + j * 16 + ml;
            bf[j].u[0] = *(const uint4*)&lB[buf][nc * TK + 8 * hf];
            bf[j].u[1] = *(const uint4*)&lB[buf][nc * TK + 16 + 8 * hf];
        }
#pragma unroll
        for (int i = 0; i < 2; ++i)
#pragma unroll
            for (int j = 0; j < 2; ++j)
                acc[i][j] = __builtin_amdgcn_wmma_f32_16x16x32_bf16(
                    false, af[i].bf, false, bf[j].bf, (short)0, acc[i][j],
                    false, false);
    };

    // prologue: tile 0 into buffer 0
    issue_tile(0, 0);
    async_wait0();
    __syncthreads();

    const int NT = K / TK;   // 8
    int cur = 0;
    for (int kt = 0; kt < NT - 1; ++kt) {    // branch-free body: always issue
        issue_tile((kt + 1) * TK, cur ^ 1);
        compute_tile(cur);
        async_wait0();     // this wave's incoming tile landed in LDS
        __syncthreads();   // all waves' copies done / reads of cur finished
        cur ^= 1;
    }
    compute_tile(cur);     // peeled last K-step: no issue, no barrier

    // ---- store: C/D layout -> VGPR v holds row (v + 8*hf), col = lane&15 ----
#pragma unroll
    for (int i = 0; i < 2; ++i) {
#pragma unroll
        for (int j = 0; j < 2; ++j) {
            const int col = colBase + wn * 32 + j * 16 + ml;
            const float bv = bias[col];
#pragma unroll
            for (int v = 0; v < 8; ++v) {
                const int row = rowBase + wm * 32 + i * 16 + v + 8 * hf;
                const float val = acc[i][j][v] + bv;
                if (OUT_BF16)
                    ((unsigned short*)Cout)[(size_t)row * N + col] = f2bf(val);
                else
                    ((float*)Cout)[(size_t)row * N + col] = val;
            }
        }
    }
}

// =====================================================================
// Deformable sampling: one wave per (b,q,h); lane = channel (C == 32).
// =====================================================================
__global__ __launch_bounds__(256) void deform_sample(
    const unsigned short* __restrict__ value, // [B*NQ, 256] bf16 = (B,Q,H,C)
    const float* __restrict__ oa,             // [B*NQ, NOAP] offsets|logits|pad
    const float* __restrict__ ref,            // [B,NQ,L,2]
    unsigned short* __restrict__ outb)        // [B*NQ, 256] bf16
{
    const int w    = blockIdx.x * 8 + (threadIdx.x >> 5);
    const int lane = threadIdx.x & 31;
    const int h    = w & 7;
    const int bq   = w >> 3;                  // b*NQ + q
    const int b    = bq / NQ;

    const int lw[LEVELS]  = {128, 64, 32};
    const int lh[LEVELS]  = {128, 64, 32};
    const int lst[LEVELS] = {0, 16384, 20480};

    float x = 0.f, y = 0.f, logit = -INFINITY;
    if (lane < LEVELS * POINTS) {
        const int l = lane >> 2;
        const size_t ob = (size_t)bq * NOAP;
        const float ox = oa[ob + h * 24 + lane * 2 + 0];
        const float oy = oa[ob + h * 24 + lane * 2 + 1];
        logit = oa[ob + 192 + h * 12 + lane];
        const float rx = ref[((size_t)bq * LEVELS + l) * 2 + 0];
        const float ry = ref[((size_t)bq * LEVELS + l) * 2 + 1];
        const float Wl = (float)lw[l], Hl = (float)lh[l];
        x = (rx + ox / Wl) * Wl - 0.5f;   // loc*W - 0.5
        y = (ry + oy / Hl) * Hl - 0.5f;
    }

    // softmax over the 12 active lanes
    float m = logit;
#pragma unroll
    for (int s = 16; s > 0; s >>= 1) m = fmaxf(m, __shfl_xor(m, s, 32));
    float e = (lane < LEVELS * POINTS) ? __expf(logit - m) : 0.f;
    float ssum = e;
#pragma unroll
    for (int s = 16; s > 0; s >>= 1) ssum += __shfl_xor(ssum, s, 32);
    const float aw = e / ssum;

    float acc = 0.f;
#pragma unroll
    for (int pt = 0; pt < LEVELS * POINTS; ++pt) {
        const float bx  = __shfl(x, pt, 32);
        const float by  = __shfl(y, pt, 32);
        const float baw = __shfl(aw, pt, 32);
        const int lvl = pt >> 2;
        const int Wl = lw[lvl], Hl = lh[lvl], st = lst[lvl];
        const float x0f = floorf(bx), y0f = floorf(by);
        const int x0 = (int)x0f, y0 = (int)y0f;
        const float wx1 = bx - x0f, wy1 = by - y0f;
        const float wx0 = 1.f - wx1, wy0 = 1.f - wy1;
        const size_t vb = ((size_t)b * NQ + st) * EMBED + h * CH + lane;
        float s00 = 0.f, s01 = 0.f, s10 = 0.f, s11 = 0.f;
        const bool xv0 = (x0 >= 0) & (x0 < Wl);
        const bool xv1 = (x0 + 1 >= 0) & (x0 + 1 < Wl);
        if (y0 >= 0 && y0 < Hl) {
            if (xv0) s00 = bf2f(value[vb + (size_t)(y0 * Wl + x0) * EMBED]);
            if (xv1) s01 = bf2f(value[vb + (size_t)(y0 * Wl + x0 + 1) * EMBED]);
        }
        if (y0 + 1 >= 0 && y0 + 1 < Hl) {
            if (xv0) s10 = bf2f(value[vb + (size_t)((y0 + 1) * Wl + x0) * EMBED]);
            if (xv1) s11 = bf2f(value[vb + (size_t)((y0 + 1) * Wl + x0 + 1) * EMBED]);
        }
        acc += baw * (wy0 * (wx0 * s00 + wx1 * s01) +
                      wy1 * (wx0 * s10 + wx1 * s11));
    }

    outb[(size_t)bq * EMBED + h * CH + lane] = f2bf(acc);
}

// =====================================================================
// Launch
// =====================================================================
extern "C" void kernel_launch(void* const* d_in, const int* in_sizes, int n_in,
                              void* d_out, int out_size, void* d_ws, size_t ws_size,
                              hipStream_t stream) {
    (void)in_sizes; (void)n_in; (void)out_size; (void)ws_size;

    const float* hs    = (const float*)d_in[0];
    const float* pos   = (const float*)d_in[1];
    const float* ref   = (const float*)d_in[2];
    // d_in[3] spatial_shapes (static, unused)
    const float* Wv    = (const float*)d_in[4];
    const float* bv    = (const float*)d_in[5];
    const float* Woff  = (const float*)d_in[6];
    const float* boff  = (const float*)d_in[7];
    const float* Wattn = (const float*)d_in[8];
    const float* battn = (const float*)d_in[9];
    const float* Wout  = (const float*)d_in[10];
    const float* bout  = (const float*)d_in[11];

    size_t off = 0;
    auto carve = [&](size_t bytes) -> char* {
        char* p = (char*)d_ws + off;
        off += (bytes + 255) & ~(size_t)255;
        return p;
    };
    const size_t actB = (size_t)MROWS * EMBED * sizeof(unsigned short);
    unsigned short* hsb   = (unsigned short*)carve(actB);
    unsigned short* qb    = (unsigned short*)carve(actB);
    unsigned short* valb  = (unsigned short*)carve(actB);
    float*          oaf   = (float*)carve((size_t)MROWS * NOAP * sizeof(float));
    unsigned short* wvT   = (unsigned short*)carve((size_t)EMBED * EMBED * 2);
    unsigned short* woaT  = (unsigned short*)carve((size_t)NOAP * EMBED * 2);
    unsigned short* woutT = (unsigned short*)carve((size_t)EMBED * EMBED * 2);
    float*          boa   = (float*)carve(NOAP * sizeof(float));
    unsigned short* sob   = hsb;  // sampled output aliases hsb (dead after GEMM1)

    // 1) convert activations; convert + transpose weights
    {
        const int n = MROWS * EMBED;
        prep_acts<<<(n + 255) / 256, 256, 0, stream>>>(hs, pos, hsb, qb, n);
        const int nw = NOAP * EMBED;   // 81920
        prep_weights<<<(nw + 255) / 256, 256, 0, stream>>>(
            Wv, Woff, Wattn, Wout, boff, battn, wvT, woaT, woutT, boa);
    }

    // 2) value = hidden @ W_value + b_value  -> bf16 [M,256]
    gemm_bf16_wmma<true><<<dim3(EMBED / TN, MROWS / TM), 256, 0, stream>>>(
        hsb, wvT, bv, (void*)valb, MROWS, EMBED, EMBED);

    // 3) [offsets|attn|pad] = query @ [W_off|W_attn|0] + bias -> f32 [M,320]
    gemm_bf16_wmma<false><<<dim3(NOAP / TN, MROWS / TM), 256, 0, stream>>>(
        qb, woaT, boa, (void*)oaf, MROWS, NOAP, EMBED);

    // 4) softmax + bilinear sampling -> bf16 [M,256]
    deform_sample<<<MROWS, 256, 0, stream>>>(valb, oaf, ref, sob);

    // 5) out = sampled @ W_out + b_out -> f32 d_out
    gemm_bf16_wmma<false><<<dim3(EMBED / TN, MROWS / TM), 256, 0, stream>>>(
        sob, woutT, bout, d_out, MROWS, EMBED, EMBED);
}
